// Model_74002286510119
// MI455X (gfx1250) — matrix operand order repaired and checked by reading, not compiled, not run on hardware
//
#include <hip/hip_runtime.h>
#include <hip/hip_bf16.h>
#include <math.h>

// ---------------- problem constants ----------------
#define BATCH   256
#define SEQ     1024
#define DIMX    128
#define DIMZ    128
#define UNITS   256
#define KDIM    512     // activation row: [ z(128) | x(128) | h(256) ]
#define NBLK    256     // 16 batch tiles x 16 unit tiles (persistent grid)
#define BLOCKT  128     // 4 wave32 waves

typedef __attribute__((ext_vector_type(16))) __bf16        v16bf;
typedef __attribute__((ext_vector_type(8)))  float         v8f;
typedef __attribute__((ext_vector_type(4)))  unsigned int  u32x4;
typedef __attribute__((ext_vector_type(8)))  int           i32x8;
typedef __attribute__((ext_vector_type(4)))  int           i32x4;

union Frag { u32x4 u[2]; v16bf v; };

__device__ __forceinline__ unsigned short f2bf(float f) {
  unsigned u = __float_as_uint(f);
  unsigned r = u + 0x7FFFu + ((u >> 16) & 1u);   // round-to-nearest-even
  return (unsigned short)(r >> 16);
}
__device__ __forceinline__ float sigm(float x)  { return 1.0f / (1.0f + __expf(-x)); }
__device__ __forceinline__ float softp(float x) { return x > 20.0f ? x : log1pf(__expf(x)); }

// Grid-wide sense barrier on a monotonically increasing counter in global ws.
__device__ __forceinline__ void grid_barrier(unsigned* bar, unsigned target) {
  __threadfence();
  __syncthreads();
  if (threadIdx.x == 0) {
    __hip_atomic_fetch_add(bar, 1u, __ATOMIC_RELEASE, __HIP_MEMORY_SCOPE_AGENT);
    while (__hip_atomic_load(bar, __ATOMIC_ACQUIRE, __HIP_MEMORY_SCOPE_AGENT) < target)
      __builtin_amdgcn_s_sleep(1);
  }
  __syncthreads();
}

extern "C" __global__ void zero_bar_kernel(unsigned* bar) { *bar = 0u; }

// ---------------- LDS layout (dynamic shared, starts at LDS address 0) ----
#define OFF_ABUF   0          // [16][512] bf16  A tile staged by TDM (16 KB)
#define OFF_BS     16384      // [64][512] bf16  gate weights, col-major (64 KB)
#define OFF_WM     81920      // [16][256] bf16  mu head (8 KB)
#define OFF_WSG    90112      // [16][256] bf16  sigma head (8 KB)
#define OFF_GBUF   98304      // 4 x [16][16] f32 gate tiles (4 KB)
#define OFF_CBUF   102400     // [16][16] f32 persistent c (1 KB)
#define OFF_MBUF   103424     // [16][16] f32 mu (1 KB)
#define OFF_SBUF   104448     // [16][16] f32 sigma-pre (1 KB)
#define SMEM_BYTES 105472

// ---------------- persistent AR-LSTM kernel ----------------
// block (bb, ub): bb = batch tile (16 rows), ub = unit tile (16 LSTM units).
// Per step: TDM stages the block's A tile [16 x 512] bf16 into LDS, 4 waves
// compute the 16x64 gate tile (i/f/g/o) with v_wmma_f32_16x16x32_bf16, the
// c-state tile lives in LDS across all 1024 steps. Blocks with ub<8 run the
// Gaussian heads for z-column tile cb=ub after the mid-step grid barrier.
extern "C" __global__ void __launch_bounds__(BLOCKT, 1)
lstm_ar_persistent(const float* __restrict__ inputs,     // [B, T, DIMX]
                   const float* __restrict__ kern,       // [256, 1024]
                   const float* __restrict__ rker,       // [256, 1024]
                   const float* __restrict__ bias,       // [1024]
                   const float* __restrict__ w_mu,       // [256, 128]
                   const float* __restrict__ b_mu,       // [128]
                   const float* __restrict__ w_sg,       // [256, 128]
                   const float* __restrict__ b_sg,       // [128]
                   const float* __restrict__ eps,        // [T-1, B, DIMZ]
                   float* __restrict__ mu_out,           // [B, T, DIMZ]
                   float* __restrict__ sg_out,           // [B, T, DIMZ]
                   unsigned short* __restrict__ act,     // [B, KDIM] bf16 bits
                   unsigned* __restrict__ bar)
{
  extern __shared__ char smem[];
  unsigned short* abuf = (unsigned short*)(smem + OFF_ABUF);
  unsigned short* Bs   = (unsigned short*)(smem + OFF_BS);
  unsigned short* Wm   = (unsigned short*)(smem + OFF_WM);
  unsigned short* Wsg  = (unsigned short*)(smem + OFF_WSG);
  float* gbuf = (float*)(smem + OFF_GBUF);
  float* cbuf = (float*)(smem + OFF_CBUF);
  float* mbuf = (float*)(smem + OFF_MBUF);
  float* sbuf = (float*)(smem + OFF_SBUF);

  const int tid  = threadIdx.x;
  const int lane = tid & 31;
  const int w    = tid >> 5;        // wave id 0..3  -> gate i/f/g/o
  const int n    = lane & 15;       // N (col) index / A-frag row index
  const int kh   = lane >> 4;       // K-half selector (ISA A/B VGPR layout)
  const int bb   = blockIdx.x & 15; // batch tile
  const int ub   = blockIdx.x >> 4; // unit tile
  const bool isHead = (ub < 8);
  const int cb   = ub;              // z-column tile for head blocks

  // ---- stage gate weights to LDS once: Bs[col][K], col = gate*16 + local unit ----
  for (int idx = tid; idx < 64 * 512; idx += BLOCKT) {
    int col = idx >> 9, K = idx & 511;
    int gcol = (col >> 4) * 256 + ub * 16 + (col & 15);
    float v = (K < 256) ? kern[K * 1024 + gcol] : rker[(K - 256) * 1024 + gcol];
    Bs[col * 512 + K] = f2bf(v);
  }
  if (isHead) {
    for (int idx = tid; idx < 16 * 256; idx += BLOCKT) {
      int col = idx >> 8, K = idx & 255;
      Wm [col * 256 + K] = f2bf(w_mu[K * 128 + cb * 16 + col]);
      Wsg[col * 256 + K] = f2bf(w_sg[K * 128 + cb * 16 + col]);
    }
  }
  // ---- init c = 0, h = 0, z = 0, x = inputs[:,0,:] ----
  for (int idx = tid; idx < 256; idx += BLOCKT) {
    cbuf[idx] = 0.0f;
    int m = idx >> 4, u = idx & 15;
    int row = bb * 16 + m;
    act[(size_t)row * KDIM + 256 + ub * 16 + u] = 0;             // h
    if (isHead) {
      int col = cb * 16 + u;
      act[(size_t)row * KDIM + col] = 0;                         // z
      act[(size_t)row * KDIM + 128 + col] =
          f2bf(inputs[((size_t)row * SEQ + 0) * DIMX + col]);    // x0
    }
  }
  unsigned epoch = 0;
  grid_barrier(bar, ++epoch * NBLK);

  const float bias_n = bias[w * 256 + ub * 16 + n];
  const float bmu_n  = isHead ? b_mu[cb * 16 + n] : 0.0f;
  const float bsg_n  = isHead ? b_sg[cb * 16 + n] : 0.0f;

  // ---- TDM descriptor for the block's A tile: contiguous 16 rows x 512 bf16 ----
  // D# group0: count=1 | lds_addr | global_addr[31:0] | global_addr[56:32],type=2
  // D# group1: data_size=2B, tensor_dim0=512, tile_dim0=512, tile_dim1=16, stride0=512
  const unsigned long long gaddr =
      (unsigned long long)(uintptr_t)(act + (size_t)(bb * 16) * KDIM);
  const u32x4 tdm_g0 = { 1u, (unsigned)OFF_ABUF,
                         (unsigned)(gaddr & 0xffffffffull),
                         (unsigned)(gaddr >> 32) | (2u << 30) };
  const i32x8 tdm_g1 = { 0x00010000,          // data_size = 1 (2 bytes)
                         (int)(512u << 16),   // tensor_dim0[15:0] in [31:16]
                         0,                   // tensor_dim0[31:16]=0, tensor_dim1 lo=0
                         (int)(512u << 16),   // tensor_dim1 hi=0, tile_dim0=512
                         16,                  // tile_dim1 = 16 rows, tile_dim2 = 0
                         512,                 // tensor_dim0_stride[31:0] = 512
                         0, 0 };
  const i32x4 tdm_z4 = { 0, 0, 0, 0 };        // groups 2/3 unused (2D tensor)
  const i32x8 tdm_z8 = { 0, 0, 0, 0, 0, 0, 0, 0 };

  const unsigned short* arow = abuf + n * KDIM;          // A-frag row (LDS)
  const unsigned short* bcol = Bs + (w * 16 + n) * 512;  // B-frag col (LDS)

  for (int t = 0; t < SEQ; ++t) {
    // ---- TDM: stage A tile [16 x 512] bf16 global -> LDS (wave 0 issues) ----
    if (tid < 32) {
      __builtin_amdgcn_tensor_load_to_lds(tdm_g0, tdm_g1, tdm_z4, tdm_z4, tdm_z8, 0);
      __builtin_amdgcn_s_wait_tensorcnt(0);
    }
    __syncthreads();

    // ================= LSTM cell: gates = [z,x,h] @ W + b =================
    v8f acc = {};
    #pragma unroll
    for (int kk = 0; kk < 16; ++kk) {
      const int kb = kk * 32;
      Frag a, b;
      // A 16x32 bf16 frag: lane kh=0 holds K {kb..kb+7, kb+16..kb+23}, kh=1 the rest
      a.u[0] = *(const u32x4*)(arow + kb + kh * 8);
      a.u[1] = *(const u32x4*)(arow + kb + 16 + kh * 8);
      // B 32x16 bf16 frag: same K order as A (kh=0: kb..kb+7, kb+16..kb+23; kh=1 the rest)
      b.u[0] = *(const u32x4*)(bcol + kb + kh * 8);
      b.u[1] = *(const u32x4*)(bcol + kb + 16 + kh * 8);
      acc = __builtin_amdgcn_wmma_f32_16x16x32_bf16(false, a.v, false, b.v,
                                                    (short)0, acc, false, false);
    }
    #pragma unroll
    for (int v = 0; v < 8; ++v)                 // C layout: M = v + 8*kh, N = n
      gbuf[w * 256 + (v + 8 * kh) * 16 + n] = acc[v] + bias_n;
    __syncthreads();

    // pointwise: c' = sig(f)*c + sig(i)*tanh(g); h = sig(o)*tanh(c')
    for (int idx = tid; idx < 256; idx += BLOCKT) {
      int m = idx >> 4, u = idx & 15;
      float gi = gbuf[0 * 256 + m * 16 + u];
      float gf = gbuf[1 * 256 + m * 16 + u];
      float gg = gbuf[2 * 256 + m * 16 + u];
      float go = gbuf[3 * 256 + m * 16 + u];
      float c  = sigm(gf) * cbuf[idx] + sigm(gi) * tanhf(gg);
      cbuf[idx] = c;
      float h  = sigm(go) * tanhf(c);
      act[(size_t)(bb * 16 + m) * KDIM + 256 + ub * 16 + u] = f2bf(h);
    }
    grid_barrier(bar, ++epoch * NBLK);

    // ================= Gaussian heads (blocks with ub < 8) =================
    if (isHead) {
      if (t + 1 < SEQ)                                          // warm next x row in L2
        __builtin_prefetch(inputs + ((size_t)(bb * 16 + n) * SEQ + (t + 1)) * DIMX, 0, 1);
      if (w < 2) {                                              // wave0 -> mu, wave1 -> sigma
        const unsigned short* hrow = act + (size_t)(bb * 16 + n) * KDIM + 256;
        const unsigned short* wcol = (w == 0 ? Wm : Wsg) + n * 256;
        v8f hacc = {};
        #pragma unroll
        for (int kk = 0; kk < 8; ++kk) {
          const int kb = kk * 32;
          Frag a, b;
          a.u[0] = *(const u32x4*)(hrow + kb + kh * 8);
          a.u[1] = *(const u32x4*)(hrow + kb + 16 + kh * 8);
          b.u[0] = *(const u32x4*)(wcol + kb + kh * 8);
          b.u[1] = *(const u32x4*)(wcol + kb + 16 + kh * 8);
          hacc = __builtin_amdgcn_wmma_f32_16x16x32_bf16(false, a.v, false, b.v,
                                                         (short)0, hacc, false, false);
        }
        float  bn   = (w == 0) ? bmu_n : bsg_n;
        float* obuf = (w == 0) ? mbuf : sbuf;
        #pragma unroll
        for (int v = 0; v < 8; ++v)
          obuf[(v + 8 * kh) * 16 + n] = hacc[v] + bn;
      }
      __syncthreads();
      for (int idx = tid; idx < 256; idx += BLOCKT) {
        int m = idx >> 4, zc = idx & 15;
        int row = bb * 16 + m, col = cb * 16 + zc;
        float mu = mbuf[m * 16 + zc];
        float sg = softp(sbuf[m * 16 + zc]) + 1e-6f;
        mu_out[((size_t)row * SEQ + t) * DIMZ + col] = mu;
        sg_out[((size_t)row * SEQ + t) * DIMZ + col] = sg;
        if (t + 1 < SEQ) {
          float e = eps[((size_t)t * BATCH + row) * DIMZ + col];
          act[(size_t)row * KDIM + col]       = f2bf(mu + sg * e);                // z_{t+1}
          act[(size_t)row * KDIM + 128 + col] =
              f2bf(inputs[((size_t)row * SEQ + (t + 1)) * DIMX + col]);           // x_{t+1}
        }
      }
    }
    grid_barrier(bar, ++epoch * NBLK);
  }
}

extern "C" void kernel_launch(void* const* d_in, const int* in_sizes, int n_in,
                              void* d_out, int out_size, void* d_ws, size_t ws_size,
                              hipStream_t stream) {
  const float* inputs = (const float*)d_in[0];
  const float* kern   = (const float*)d_in[1];
  const float* rker   = (const float*)d_in[2];
  const float* bias   = (const float*)d_in[3];
  const float* w_mu   = (const float*)d_in[4];
  const float* b_mu   = (const float*)d_in[5];
  const float* w_sg   = (const float*)d_in[6];
  const float* b_sg   = (const float*)d_in[7];
  const float* eps    = (const float*)d_in[8];

  float* mu_out = (float*)d_out;
  float* sg_out = mu_out + (size_t)BATCH * SEQ * DIMZ;

  unsigned*       bar = (unsigned*)d_ws;                          // 4 B counter
  unsigned short* act = (unsigned short*)((char*)d_ws + 256);     // 256 KB activation buffer (L2-resident)

  (void)in_sizes; (void)n_in; (void)out_size; (void)ws_size;

  (void)hipFuncSetAttribute((const void*)lstm_ar_persistent,
                            hipFuncAttributeMaxDynamicSharedMemorySize, SMEM_BYTES);

  zero_bar_kernel<<<1, 1, 0, stream>>>(bar);
  lstm_ar_persistent<<<NBLK, BLOCKT, SMEM_BYTES, stream>>>(
      inputs, kern, rker, bias, w_mu, b_mu, w_sg, b_sg, eps,
      mu_out, sg_out, act, bar);
}
